// _MockEvoformerBlock_43885975830776
// MI455X (gfx1250) — matrix operand order repaired and checked by reading, not compiled, not run on hardware
//
#include <hip/hip_runtime.h>
#include <hip/hip_bf16.h>

// ---------------------------------------------------------------------------
// Types for CDNA5 WMMA (wave32): v_wmma_f32_16x16x32_bf16
// ---------------------------------------------------------------------------
typedef __attribute__((ext_vector_type(16))) __bf16 v16bf;
typedef __attribute__((ext_vector_type(8)))  __bf16 bf16x8;
typedef __attribute__((ext_vector_type(8)))  float  v8f;
typedef __attribute__((ext_vector_type(4)))  int    v4i;

union Frag { bf16x8 h[2]; v16bf v; };

// --------------------------- async global->LDS ------------------------------
#ifdef __has_builtin
#  if __has_builtin(__builtin_amdgcn_global_load_async_to_lds_b128)
#    define HAVE_ASYNC 1
#  endif
#endif
#ifndef HAVE_ASYNC
#  define HAVE_ASYNC 0
#endif

typedef __attribute__((address_space(1))) v4i* gv4i_p;   // global-AS v4i*
typedef __attribute__((address_space(3))) v4i* lv4i_p;   // LDS-AS v4i*

__device__ __forceinline__ void async_copy16(const void* g, void* l) {
#if HAVE_ASYNC
    __builtin_amdgcn_global_load_async_to_lds_b128((gv4i_p)g, (lv4i_p)l, 0, 0);
#else
    *(int4*)l = *(const int4*)g;      // fallback: sync load + ds_store
#endif
}

__device__ __forceinline__ void wait_async_le8() {
#if HAVE_ASYNC
    asm volatile("s_wait_asynccnt 8" ::: "memory");
#endif
}
__device__ __forceinline__ void wait_async_le4() {
#if HAVE_ASYNC
    asm volatile("s_wait_asynccnt 4" ::: "memory");
#endif
}
__device__ __forceinline__ void wait_async_le0() {
#if HAVE_ASYNC
    asm volatile("s_wait_asynccnt 0" ::: "memory");
#endif
}

__device__ __forceinline__ __bf16 f2bf(float f) {
    union { float f; unsigned u; } c; c.f = f;
    unsigned u = c.u;
    unsigned r = u + 0x7FFFu + ((u >> 16) & 1u);   // RNE fp32->bf16
    union { unsigned short s; __bf16 b; } o; o.s = (unsigned short)(r >> 16);
    return o.b;
}

// ---------------------------------------------------------------------------
// fp32 -> bf16 elementwise convert
// ---------------------------------------------------------------------------
__global__ void cvt_f32_bf16(const float* __restrict__ in, __bf16* __restrict__ out, int n) {
    int i = blockIdx.x * blockDim.x + threadIdx.x;
    if (i < n) out[i] = f2bf(in[i]);
}

// ---------------------------------------------------------------------------
// GEMM: C[M,N] = A[M,K](bf16 rm) @ W[N,K](bf16 rm)^T (+bias) (+ReLU), batched.
// outMode: 0 = bf16 row-major, 1 = f32 row-major, 2 = bf16 transposed C[n,m]
//
// Block (8 waves) computes a 128x128 C tile. A/B k-slices (128x32 bf16) are
// TRIPLE-buffered in LDS, filled with async global->LDS b128 copies (ASYNCcnt,
// 2 fills in flight) while WMMAs consume an older buffer. Per k-step each wave
// loads 1 A-fragment + all 8 B-fragments behind ONE s_wait_dscnt, then fires
// 8 back-to-back WMMAs. LDS rows padded to 80B to avoid ds bank conflicts.
// ---------------------------------------------------------------------------
#define ROW_BYTES 80              // 64B data (32 bf16) + 16B pad
#define TILE_ROWS 128
#define STAGE_BYTES (2 * TILE_ROWS * ROW_BYTES)   // A-half + B-half

__global__ void __launch_bounds__(256)
gemm_bf16_wmma(const __bf16* __restrict__ A,
               const __bf16* __restrict__ W,
               const float*  __restrict__ bias,
               void* __restrict__ Cout,
               int M, int N, int K,
               long long strideA, long long strideW, long long strideC,
               int ldc, int outMode, int relu) {
    __shared__ __align__(16) char lds[3 * STAGE_BYTES];

    const int tid  = threadIdx.x;
    const int lane = tid & 31;
    const int wv   = tid >> 5;            // wave 0..7
    const int tn   = blockIdx.x;          // N/128
    const int tm   = blockIdx.y;          // M/128
    const int b    = blockIdx.z;

    A += (long long)b * strideA + (size_t)(tm * 128) * K;
    W += (long long)b * strideW + (size_t)(tn * 128) * K;

    // fill mapping: 32 lanes = 8 rows x 4 16B-chunks; wave w fills rows 16w..16w+15
    const int frow = lane >> 2;
    const int fchk = lane & 3;
    const int row0 = wv * 16 + frow;          // rows row0 and row0+8
    const int row1 = row0 + 8;

    const char* gA0 = (const char*)(A + (size_t)row0 * K) + fchk * 16;
    const char* gA1 = (const char*)(A + (size_t)row1 * K) + fchk * 16;
    const char* gW0 = (const char*)(W + (size_t)row0 * K) + fchk * 16;
    const char* gW1 = (const char*)(W + (size_t)row1 * K) + fchk * 16;
    const int lA0 = row0 * ROW_BYTES + fchk * 16;
    const int lA1 = row1 * ROW_BYTES + fchk * 16;
    const int lB0 = TILE_ROWS * ROW_BYTES + lA0;
    const int lB1 = TILE_ROWS * ROW_BYTES + lA1;

    auto issue_fill = [&](int buf) {
        char* base = lds + buf * STAGE_BYTES;
        async_copy16(gA0, base + lA0);
        async_copy16(gW0, base + lB0);
        async_copy16(gA1, base + lA1);
        async_copy16(gW1, base + lB1);
        gA0 += 64; gA1 += 64; gW0 += 64; gW1 += 64;   // advance one k-step
    };

    const int l  = lane & 15;
    const int hi = lane >> 4;
    const int aOff = (wv * 16 + l) * ROW_BYTES + hi * 16;
    const int bOff = TILE_ROWS * ROW_BYTES + l * ROW_BYTES + hi * 16;

    v8f acc[8];
#pragma unroll
    for (int j = 0; j < 8; ++j) acc[j] = (v8f){0.f,0.f,0.f,0.f,0.f,0.f,0.f,0.f};

    auto compute = [&](int buf) {
        const char* base = lds + buf * STAGE_BYTES;
        Frag af;
        af.h[0] = *(const bf16x8*)(base + aOff);
        af.h[1] = *(const bf16x8*)(base + aOff + 32);
        Frag bfr[8];
#pragma unroll
        for (int j = 0; j < 8; ++j) {
            const char* bb = base + bOff + j * 16 * ROW_BYTES;
            bfr[j].h[0] = *(const bf16x8*)(bb);
            bfr[j].h[1] = *(const bf16x8*)(bb + 32);
        }
#pragma unroll
        for (int j = 0; j < 8; ++j)
            acc[j] = __builtin_amdgcn_wmma_f32_16x16x32_bf16(
                false, af.v, false, bfr[j].v, (short)0, acc[j], false, false);
    };

    const int KT = K >> 5;                // k-steps of 32 (>= 2 here)
    issue_fill(0);
    issue_fill(1);

    int br = 0;                           // buffer to read
    int bw = 2;                           // buffer to write (kt+2)
    for (int kt = 0; kt + 2 < KT; ++kt) {
        issue_fill(bw);
        wait_async_le8();                 // fills for chunk kt complete
        __syncthreads();                  // ... and visible block-wide
        compute(br);
        __syncthreads();                  // reads done before buffer reuse
        br = (br == 2) ? 0 : br + 1;
        bw = (bw == 2) ? 0 : bw + 1;
    }
    // kt = KT-2
    wait_async_le4();
    __syncthreads();
    compute(br);
    __syncthreads();
    br = (br == 2) ? 0 : br + 1;
    // kt = KT-1
    wait_async_le0();
    __syncthreads();
    compute(br);

    // ---- store: VGPR i -> row m0+i; lanes 16-31 -> rows +8; N = lane&15 ----
    const int m0 = tm * 128 + wv * 16 + hi * 8;
    const int n_base = tn * 128 + l;
#pragma unroll
    for (int j = 0; j < 8; ++j) {
        const int n = n_base + j * 16;
        const float bv = bias ? bias[n] : 0.0f;
        float v[8];
#pragma unroll
        for (int i = 0; i < 8; ++i) {
            float x = acc[j][i] + bv;
            v[i] = relu ? fmaxf(x, 0.0f) : x;
        }
        if (outMode == 0) {
#pragma unroll
            for (int i = 0; i < 8; ++i)
                ((__bf16*)Cout)[(long long)b * strideC + (size_t)(m0 + i) * ldc + n] = f2bf(v[i]);
        } else if (outMode == 1) {
#pragma unroll
            for (int i = 0; i < 8; ++i)
                ((float*)Cout)[(long long)b * strideC + (size_t)(m0 + i) * ldc + n] = v[i];
        } else {                          // transposed bf16: contiguous in m -> 16B store
            bf16x8 pk;
#pragma unroll
            for (int i = 0; i < 8; ++i) pk[i] = f2bf(v[i]);
            *(bf16x8*)&((__bf16*)Cout)[(long long)b * strideC + (size_t)n * ldc + m0] = pk;
        }
    }
}

// ---------------------------------------------------------------------------
// Row softmax: one 256-thread block per row of length n. S (f32) -> P (bf16).
// ---------------------------------------------------------------------------
__global__ void softmax_rows(const float* __restrict__ S, __bf16* __restrict__ P, int n) {
    const int row = blockIdx.x;
    const float* s = S + (size_t)row * n;
    __bf16*      p = P + (size_t)row * n;
    const int tid = threadIdx.x;
    __shared__ float red[8];

    float m = -3.0e38f;
    for (int i = tid; i < n; i += 256) m = fmaxf(m, s[i]);
#pragma unroll
    for (int off = 16; off > 0; off >>= 1) m = fmaxf(m, __shfl_xor(m, off, 32));
    if ((tid & 31) == 0) red[tid >> 5] = m;
    __syncthreads();
    float mm = red[0];
#pragma unroll
    for (int i = 1; i < 8; ++i) mm = fmaxf(mm, red[i]);
    __syncthreads();

    float sum = 0.0f;
    for (int i = tid; i < n; i += 256) sum += __expf(s[i] - mm);
#pragma unroll
    for (int off = 16; off > 0; off >>= 1) sum += __shfl_xor(sum, off, 32);
    if ((tid & 31) == 0) red[tid >> 5] = sum;
    __syncthreads();
    float tot = 0.0f;
#pragma unroll
    for (int i = 0; i < 8; ++i) tot += red[i];
    const float inv = 1.0f / tot;

    for (int i = tid; i < n; i += 256) p[i] = f2bf(__expf(s[i] - mm) * inv);
}

// ---------------------------------------------------------------------------
// Host-side orchestration
// ---------------------------------------------------------------------------
extern "C" void kernel_launch(void* const* d_in, const int* in_sizes, int n_in,
                              void* d_out, int out_size, void* d_ws, size_t ws_size,
                              hipStream_t stream) {
    (void)in_sizes; (void)n_in; (void)out_size; (void)ws_size;
    constexpr int B = 4, S = 2048, D = 512, H = 2048;   // H = 4*D
    constexpr int M = B * S;                            // 8192

    const float* x  = (const float*)d_in[0];
    const float* Wq = (const float*)d_in[1];
    const float* bq = (const float*)d_in[2];
    const float* Wk = (const float*)d_in[3];
    const float* bk = (const float*)d_in[4];
    const float* Wv = (const float*)d_in[5];
    const float* bv = (const float*)d_in[6];
    const float* W1 = (const float*)d_in[7];
    const float* b1 = (const float*)d_in[8];
    const float* W2 = (const float*)d_in[9];
    const float* b2 = (const float*)d_in[10];

    char* ws = (char*)d_ws;
    size_t off = 0;
    auto take = [&](size_t bytes) -> char* {
        char* p = ws + off;
        off = (off + bytes + 255) & ~(size_t)255;
        return p;
    };

    __bf16* x_bf  = (__bf16*)take((size_t)M * D * 2);
    __bf16* wq_bf = (__bf16*)take((size_t)D * D * 2);
    __bf16* wk_bf = (__bf16*)take((size_t)D * D * 2);
    __bf16* wv_bf = (__bf16*)take((size_t)D * D * 2);
    __bf16* w1_bf = (__bf16*)take((size_t)H * D * 2);
    __bf16* w2_bf = (__bf16*)take((size_t)D * H * 2);
    __bf16* q_bf  = (__bf16*)take((size_t)M * D * 2);
    __bf16* k_bf  = (__bf16*)take((size_t)M * D * 2);
    __bf16* vT_bf = (__bf16*)take((size_t)B * D * S * 2);  // per batch: [D][S]
    float*  scor  = (float*) take((size_t)B * S * S * 4);
    __bf16* P_bf  = (__bf16*)take((size_t)B * S * S * 2);
    __bf16* at_bf = (__bf16*)take((size_t)M * D * 2);
    __bf16* h_bf  = (__bf16*)take((size_t)M * H * 2);

    auto cvt = [&](const float* in, __bf16* out, int n) {
        cvt_f32_bf16<<<(n + 255) / 256, 256, 0, stream>>>(in, out, n);
    };
    cvt(x,  x_bf,  M * D);
    cvt(Wq, wq_bf, D * D);
    cvt(Wk, wk_bf, D * D);
    cvt(Wv, wv_bf, D * D);
    cvt(W1, w1_bf, H * D);
    cvt(W2, w2_bf, D * H);

    auto gemm = [&](const __bf16* A, const __bf16* Wm, const float* bias, void* C,
                    int Mg, int Ng, int Kg,
                    long long sA, long long sW, long long sC,
                    int ldc, int outMode, int relu, int batch) {
        dim3 grid(Ng / 128, Mg / 128, batch);
        gemm_bf16_wmma<<<grid, 256, 0, stream>>>(A, Wm, bias, C, Mg, Ng, Kg,
                                                 sA, sW, sC, ldc, outMode, relu);
    };

    // q = x @ Wq^T + bq    [8192,512] bf16
    gemm(x_bf, wq_bf, bq, q_bf, M, D, D, 0, 0, 0, D, /*bf16*/0, 0, 1);
    // k = x @ Wk^T + bk
    gemm(x_bf, wk_bf, bk, k_bf, M, D, D, 0, 0, 0, D, 0, 0, 1);
    // vT[b][d][t] = (x @ Wv^T + bv)^T : batched, transposed store
    gemm(x_bf, wv_bf, bv, vT_bf, S, D, D,
         (long long)S * D, 0, (long long)D * S, S, /*bf16-T*/2, 0, B);
    // scores[b] = q[b] @ k[b]^T   (f32)
    gemm(q_bf, k_bf, nullptr, scor, S, S, D,
         (long long)S * D, (long long)S * D, (long long)S * S, S, /*f32*/1, 0, B);
    // P = softmax(scores) (bf16)
    softmax_rows<<<B * S, 256, 0, stream>>>(scor, P_bf, S);
    // attn[b] = P[b] @ v[b]  (B-matrix rows come from vT)
    gemm(P_bf, vT_bf, nullptr, at_bf, S, D, S,
         (long long)S * S, (long long)D * S, (long long)S * D, D, 0, 0, B);
    // h = relu(attn @ W1^T + b1)   [8192,2048] bf16
    gemm(at_bf, w1_bf, b1, h_bf, M, H, D, 0, 0, 0, H, 0, /*relu*/1, 1);
    // out = h @ W2^T + b2          [8192,512] f32 -> d_out
    gemm(h_bf, w2_bf, b2, d_out, M, D, H, 0, 0, 0, D, /*f32*/1, 0, 1);
}
